// MoELayer_80788334838412
// MI455X (gfx1250) — matrix operand order repaired and checked by reading, not compiled, not run on hardware
//
#include <hip/hip_runtime.h>
#include <hip/hip_bf16.h>
#include <stdint.h>

// Problem constants: B=2, S=4096 -> T=8192 tokens
#define Tt 8192
#define Dd 1024
#define Ee 8
#define Hh 2816
#define HC 704            // H chunk held in LDS (Hh/4)
#define NHC 4

typedef __attribute__((ext_vector_type(16))) __bf16 v16bf;
typedef __attribute__((ext_vector_type(8)))  __bf16 v8bf;
typedef __attribute__((ext_vector_type(8)))  float  v8f;

// ---------------------------------------------------------------------------
// CDNA5 async global->LDS copy (ASYNCcnt-tracked DMA path)
// ---------------------------------------------------------------------------
__device__ __forceinline__ void async_copy_b128(unsigned lds_off, const void* gptr) {
    asm volatile("global_load_async_to_lds_b128 %0, %1, off"
                 :: "v"(lds_off), "v"((unsigned long long)(uintptr_t)gptr)
                 : "memory");
}
__device__ __forceinline__ void wait_asynccnt0() {
    asm volatile("s_wait_asynccnt 0x0" ::: "memory");
}

// ---------------------------------------------------------------------------
// fp32 -> bf16 elementwise convert (for x), 8 elems/thread
// ---------------------------------------------------------------------------
__global__ __launch_bounds__(256) void cvt_bf16_kernel(const float* __restrict__ src,
                                                       __bf16* __restrict__ dst) {
    long i = ((long)blockIdx.x * 256 + threadIdx.x) * 8;
    float4 a = *(const float4*)(src + i);
    float4 b = *(const float4*)(src + i + 4);
    v8bf o;
    o[0] = (__bf16)a.x; o[1] = (__bf16)a.y; o[2] = (__bf16)a.z; o[3] = (__bf16)a.w;
    o[4] = (__bf16)b.x; o[5] = (__bf16)b.y; o[6] = (__bf16)b.z; o[7] = (__bf16)b.w;
    *(v8bf*)(dst + i) = o;
}

// ---------------------------------------------------------------------------
// fp32 [R][C] -> bf16 [C][R] tiled transpose-convert (per blockIdx.z matrix)
// R, C multiples of 64.
// ---------------------------------------------------------------------------
__global__ __launch_bounds__(256) void transpose_cvt_kernel(const float* __restrict__ src,
                                                            __bf16* __restrict__ dst,
                                                            int R, int C) {
    __shared__ __bf16 tile[64][72];  // padded
    const int bx = blockIdx.x * 64;  // col base in src
    const int by = blockIdx.y * 64;  // row base in src
    const float* s = src + (size_t)blockIdx.z * (size_t)R * C;
    __bf16* d = dst + (size_t)blockIdx.z * (size_t)R * C;
    const int tx = threadIdx.x & 63, ty = threadIdx.x >> 6;
    for (int r = ty; r < 64; r += 4)
        tile[r][tx] = (__bf16)s[(size_t)(by + r) * C + bx + tx];
    __syncthreads();
    for (int r = ty; r < 64; r += 4)
        d[(size_t)(bx + r) * R + by + tx] = tile[tx][r];
}

// ---------------------------------------------------------------------------
// Router: one wave per token; logits via LDS-resident router weights,
// wave32 shuffle reduction, top-2 softmax, atomic compaction into lists.
// ---------------------------------------------------------------------------
__global__ __launch_bounds__(256) void moe_router_kernel(
    const float* __restrict__ x, const float* __restrict__ rw,
    int* __restrict__ counts, int* __restrict__ tok_idx, float* __restrict__ tok_w)
{
    __shared__ float rws[Ee * Dd];
    const int tid = threadIdx.x;
    for (int i = tid; i < Ee * Dd; i += 256) rws[i] = rw[i];
    __syncthreads();

    const int wave = tid >> 5, lane = tid & 31;
    const int t = blockIdx.x * 8 + wave;

    float acc[Ee];
#pragma unroll
    for (int e = 0; e < Ee; ++e) acc[e] = 0.0f;
    for (int d = lane; d < Dd; d += 32) {
        float xv = x[(long)t * Dd + d];
#pragma unroll
        for (int e = 0; e < Ee; ++e) acc[e] += xv * rws[e * Dd + d];
    }
#pragma unroll
    for (int e = 0; e < Ee; ++e) {
#pragma unroll
        for (int off = 16; off > 0; off >>= 1)
            acc[e] += __shfl_xor(acc[e], off, 32);
    }
    if (lane == 0) {
        int b0 = 0; float v0 = acc[0];
#pragma unroll
        for (int e = 1; e < Ee; ++e) if (acc[e] > v0) { v0 = acc[e]; b0 = e; }
        int b1 = (b0 == 0) ? 1 : 0; float v1 = acc[b1];
#pragma unroll
        for (int e = 0; e < Ee; ++e)
            if (e != b0 && acc[e] > v1) { v1 = acc[e]; b1 = e; }
        float e1 = __expf(v1 - v0);
        float inv = 1.0f / (1.0f + e1);
        int p0 = atomicAdd(&counts[b0], 1);
        tok_idx[b0 * Tt + p0] = t; tok_w[b0 * Tt + p0] = inv;
        int p1 = atomicAdd(&counts[b1], 1);
        tok_idx[b1 * Tt + p1] = t; tok_w[b1 * Tt + p1] = e1 * inv;
    }
}

// ---------------------------------------------------------------------------
// Fused per-expert SwiGLU FFN, 64 tokens per block, 8 waves.
// out^T = W^T x X^T formulation: weights (pre-transposed bf16) are the WMMA A
// operand streamed straight from global in fragment layout; activations are
// the B operand with contiguous per-lane K runs from LDS. No staging, no
// transposes, barrier-free K loops.
//   LDS: Xs[64][1024] bf16 (128KB) + Hs[64][704] bf16 chunk (88KB) + lists.
// ---------------------------------------------------------------------------
#define XS_BYTES  (64 * Dd * 2)     // 131072
#define HSC_BYTES (64 * HC * 2)     // 90112
#define TK_BYTES  (64 * 4 * 2)      // 512
#define SMEM_BYTES (XS_BYTES + HSC_BYTES + TK_BYTES)  // 221696

__global__ __launch_bounds__(256) void moe_ffn_kernel(
    const __bf16* __restrict__ xb,   // [T][D]
    const __bf16* __restrict__ w1t,  // [E][H][D]
    const __bf16* __restrict__ w3t,  // [E][H][D]
    const __bf16* __restrict__ w2t,  // [E][D][H]
    const int* __restrict__ counts, const int* __restrict__ tok_idx,
    const float* __restrict__ tok_w, float* __restrict__ out)
{
    extern __shared__ char smem[];
    __bf16* Xs = (__bf16*)smem;                          // [64][1024]
    __bf16* Hs = (__bf16*)(smem + XS_BYTES);             // [64][704] (chunk)
    int*    tok_s = (int*)(smem + XS_BYTES + HSC_BYTES);
    float*  wt_s  = (float*)(tok_s + 64);

    const int e = blockIdx.y, mtile = blockIdx.x;
    const int n_e = counts[e];
    if (mtile * 64 >= n_e) return;   // uniform: EXEC stays all-ones for WMMA

    const int tid = threadIdx.x, wave = tid >> 5, lane = tid & 31;
    const int lrow = lane & 15, lhi = lane >> 4;

    if (tid < 64) {
        int r = mtile * 64 + tid;
        bool valid = r < n_e;
        tok_s[tid] = valid ? tok_idx[e * Tt + r] : 0;
        wt_s [tid] = valid ? tok_w [e * Tt + r] : 0.0f;  // 0 => no contribution
    }
    __syncthreads();

    // Async DMA gather of 64 token rows (bf16) into Xs: 16B chunks.
    {
        const unsigned xs_base = (unsigned)(uintptr_t)Xs;
        for (int c = tid; c < 64 * (Dd / 8); c += 256) {
            int r = c >> 7, col = (c & 127) << 3;
            async_copy_b128(xs_base + (unsigned)((r * Dd + col) * 2),
                            xb + (long)tok_s[r] * Dd + col);
        }
        wait_asynccnt0();
    }
    __syncthreads();

    const __bf16* w1e = w1t + (long)e * Hh * Dd;
    const __bf16* w3e = w3t + (long)e * Hh * Dd;
    const __bf16* w2e = w2t + (long)e * Dd * Hh;

    for (int hc = 0; hc < NHC; ++hc) {
        // ---- Phase 1: Hs[token][0..704) = silu(X W1) .* (X W3) for this chunk
        for (int ht = wave; ht < HC / 16; ht += 8) {
            const int hrow = hc * HC + ht * 16 + lrow;   // global H row (A operand)
            const __bf16* a1p = w1e + (long)hrow * Dd + lhi * 8;
            const __bf16* a3p = w3e + (long)hrow * Dd + lhi * 8;
            v8f acc1[4] = {}; v8f acc3[4] = {};
            for (int k = 0; k < Dd; k += 32) {
                __builtin_prefetch((const void*)(a1p + k + 64), 0, 1);
                v8bf t0 = *(const v8bf*)(a1p + k);
                v8bf t1 = *(const v8bf*)(a1p + k + 16);
                v8bf u0 = *(const v8bf*)(a3p + k);
                v8bf u1 = *(const v8bf*)(a3p + k + 16);
                v16bf a1, a3;
#pragma unroll
                for (int i = 0; i < 8; ++i) {
                    a1[i] = t0[i]; a1[8 + i] = t1[i];
                    a3[i] = u0[i]; a3[8 + i] = u1[i];
                }
#pragma unroll
                for (int mt = 0; mt < 4; ++mt) {
                    v8bf b0 = *(const v8bf*)&Xs[(mt * 16 + lrow) * Dd + k + lhi * 8];
                    v8bf b1 = *(const v8bf*)&Xs[(mt * 16 + lrow) * Dd + k + 16 + lhi * 8];
                    v16bf b = __builtin_shufflevector(b0, b1, 0, 1, 2, 3, 4, 5, 6, 7, 8, 9, 10, 11, 12, 13, 14, 15);
                    acc1[mt] = __builtin_amdgcn_wmma_f32_16x16x32_bf16(
                        false, a1, false, b, (short)0, acc1[mt], false, false);
                    acc3[mt] = __builtin_amdgcn_wmma_f32_16x16x32_bf16(
                        false, a3, false, b, (short)0, acc3[mt], false, false);
                }
            }
            // SwiGLU; C-tile lanes own 8 consecutive H values -> one b128 store
#pragma unroll
            for (int mt = 0; mt < 4; ++mt) {
                v8bf hv;
#pragma unroll
                for (int v = 0; v < 8; ++v) {
                    float g = acc1[mt][v];
                    hv[v] = (__bf16)((g / (1.0f + __expf(-g))) * acc3[mt][v]);
                }
                *(v8bf*)&Hs[(mt * 16 + lrow) * HC + ht * 16 + lhi * 8] = hv;
            }
        }
        __syncthreads();

        // ---- Phase 2: out += (Hs @ W2[chunk rows]) * routing weight
        for (int dt = wave; dt < Dd / 16; dt += 8) {
            const __bf16* a2p = w2e + (long)(dt * 16 + lrow) * Hh + hc * HC + lhi * 8;
            v8f acc[4] = {};
            for (int k = 0; k < HC; k += 32) {
                __builtin_prefetch((const void*)(a2p + k + 64), 0, 1);
                v8bf t0 = *(const v8bf*)(a2p + k);
                v8bf t1 = *(const v8bf*)(a2p + k + 16);
                v16bf a;
#pragma unroll
                for (int i = 0; i < 8; ++i) { a[i] = t0[i]; a[8 + i] = t1[i]; }
#pragma unroll
                for (int mt = 0; mt < 4; ++mt) {
                    v8bf b0 = *(const v8bf*)&Hs[(mt * 16 + lrow) * HC + k + lhi * 8];
                    v8bf b1 = *(const v8bf*)&Hs[(mt * 16 + lrow) * HC + k + 16 + lhi * 8];
                    v16bf b = __builtin_shufflevector(b0, b1, 0, 1, 2, 3, 4, 5, 6, 7, 8, 9, 10, 11, 12, 13, 14, 15);
                    acc[mt] = __builtin_amdgcn_wmma_f32_16x16x32_bf16(
                        false, a, false, b, (short)0, acc[mt], false, false);
                }
            }
            // C-tile: M = D index (dt*16 + v + 8*lhi), N = token (mt*16 + lrow)
#pragma unroll
            for (int mt = 0; mt < 4; ++mt) {
                int r = mt * 16 + lrow;
                float wgt = wt_s[r];
                long obase = (long)tok_s[r] * Dd + dt * 16 + lhi * 8;
#pragma unroll
                for (int v = 0; v < 8; ++v)
                    atomicAdd(&out[obase + v], acc[mt][v] * wgt);
            }
        }
        __syncthreads();  // Hs reused next chunk
    }
}

// ---------------------------------------------------------------------------
// Host-side launch
// ---------------------------------------------------------------------------
extern "C" void kernel_launch(void* const* d_in, const int* in_sizes, int n_in,
                              void* d_out, int out_size, void* d_ws, size_t ws_size,
                              hipStream_t stream) {
    const float* x  = (const float*)d_in[0];
    const float* rw = (const float*)d_in[1];
    const float* w1 = (const float*)d_in[2];
    const float* w2 = (const float*)d_in[3];
    const float* w3 = (const float*)d_in[4];
    float* out = (float*)d_out;
    (void)in_sizes; (void)n_in; (void)ws_size;

    char* ws = (char*)d_ws;
    size_t off = 0;
    int*   counts  = (int*)(ws + off);  off += 256;
    int*   tok_idx = (int*)(ws + off);  off += (size_t)Ee * Tt * 4;
    float* tok_w   = (float*)(ws + off); off += (size_t)Ee * Tt * 4;
    off = (off + 255) & ~(size_t)255;
    __bf16* xb  = (__bf16*)(ws + off);  off += (size_t)Tt * Dd * 2;
    __bf16* w1t = (__bf16*)(ws + off);  off += (size_t)Ee * Hh * Dd * 2;
    __bf16* w3t = (__bf16*)(ws + off);  off += (size_t)Ee * Hh * Dd * 2;
    __bf16* w2t = (__bf16*)(ws + off);  off += (size_t)Ee * Hh * Dd * 2;

    // Atomic-accumulated output: zero every launch (graph replays!).
    hipMemsetAsync(d_out, 0, (size_t)out_size * sizeof(float), stream);
    hipMemsetAsync(counts, 0, 256, stream);

    // One-time (per launch) bf16 conversion / transposition of operands.
    cvt_bf16_kernel<<<(Tt * Dd) / 2048, 256, 0, stream>>>(x, xb);
    transpose_cvt_kernel<<<dim3(Hh / 64, Dd / 64, Ee), 256, 0, stream>>>(w1, w1t, Dd, Hh);
    transpose_cvt_kernel<<<dim3(Hh / 64, Dd / 64, Ee), 256, 0, stream>>>(w3, w3t, Dd, Hh);
    transpose_cvt_kernel<<<dim3(Dd / 64, Hh / 64, Ee), 256, 0, stream>>>(w2, w2t, Hh, Dd);

    moe_router_kernel<<<Tt / 8, 256, 0, stream>>>(x, rw, counts, tok_idx, tok_w);

    hipFuncSetAttribute(reinterpret_cast<const void*>(moe_ffn_kernel),
                        hipFuncAttributeMaxDynamicSharedMemorySize, SMEM_BYTES);
    moe_ffn_kernel<<<dim3(Tt / 64, Ee), 256, SMEM_BYTES, stream>>>(
        xb, w1t, w3t, w2t, counts, tok_idx, tok_w, out);
}
